// DilatedAttention_15135464751385
// MI455X (gfx1250) — compile-verified, matrix operand order checked by reading, not run on hardware
//
#include <hip/hip_runtime.h>

// DilatedAttention for (b,h,s,d)=(2,24,8192,64), W=[64,128,256], R=[1,2,4].
// Each segment is an independent m x m SDPA (m = 64/63/62) over d=64.
// Memory-bound (~403 MB total traffic @ 23.3 TB/s ~= 17us), so: 1 workgroup
// per segment, single pass over Q/K/V, f16 WMMA (f32 accumulate), and each
// workgroup zero-fills the non-dilated rows of its own window (windows tile
// the sequence exactly, so no global memset pass is needed).

typedef __attribute__((ext_vector_type(16))) _Float16 v16h;
typedef __attribute__((ext_vector_type(8)))  _Float16 v8h;
typedef __attribute__((ext_vector_type(4)))  _Float16 v4h;
typedef __attribute__((ext_vector_type(8)))  float    v8f;

#define LDH 64   // row stride (halves) of all LDS tiles

// Load a 16x32 f16 A/B fragment from a row-major LDS tile (ld = 64 halves).
// ISA layout: lane L -> row row0+(L&15); elems 0..7 = K[kbase + 8*(L>>4) .. +7],
// elems 8..15 = K[kbase + 16 + 8*(L>>4) .. +7]  -> two ds_load_b128.
__device__ __forceinline__ v16h frag_ld(const _Float16* __restrict__ smem,
                                        int row0, int kbase, int lane) {
  const int r = row0 + (lane & 15);
  const int h = (lane >> 4) << 3;
  v8h lo = *(const v8h*)(smem + r * LDH + kbase + h);
  v8h hi = *(const v8h*)(smem + r * LDH + kbase + 16 + h);
  return __builtin_shufflevector(lo, hi, 0, 1, 2, 3, 4, 5, 6, 7,
                                         8, 9, 10, 11, 12, 13, 14, 15);
}

__global__ __launch_bounds__(128)
void dilated_attn_f16wmma(const float* __restrict__ q,
                          const float* __restrict__ k,
                          const float* __restrict__ v,
                          float* __restrict__ out) {
  __shared__ _Float16 sQ[64 * LDH];
  __shared__ _Float16 sK[64 * LDH];
  __shared__ _Float16 sVt[64 * LDH];   // transposed: sVt[d][kk]
  __shared__ _Float16 sP[64 * LDH];

  const int tid  = threadIdx.x;
  const int lane = tid & 31;
  const int wv   = tid >> 5;

  // ---- decode (group, b, head, segment) from blockIdx ----
  int bid = blockIdx.x;
  int grp, w, r, rsh, off, mv, nseg;
  if (bid < 2048)      { grp = 0; w = 64;  r = 1; rsh = 0; off = 0; mv = 64; nseg = 128; }
  else if (bid < 3072) { grp = 1; w = 128; r = 2; rsh = 1; off = 2; mv = 63; nseg = 64; bid -= 2048; }
  else                 { grp = 2; w = 256; r = 4; rsh = 2; off = 8; mv = 62; nseg = 32; bid -= 3072; }

  const int seg  = bid % nseg;
  const int rest = bid / nseg;
  const int hg   = rest & 7;
  const int bb   = rest >> 3;
  const int head = grp * 8 + hg;

  const int winBase = (((bb * 24 + head) * 8192) + seg * w) * 64; // window start (floats)
  const int base    = winBase + off * 64;                          // dilated row 0

  // ---- stage Q,K row-major f16 and V transposed into LDS (zero-pad rows >= mv) ----
  #pragma unroll
  for (int u = 0; u < 8; ++u) {
    const int e   = tid + u * 128;        // 0..1023 -> 64 rows x 16 float4
    const int row = e >> 4;
    const int c4  = (e & 15) << 2;
    if (row < mv) {
      const int g = base + row * (r * 64) + c4;
      float4 qv = *(const float4*)(q + g);
      float4 kv = *(const float4*)(k + g);
      float4 vv = *(const float4*)(v + g);
      v4h hq = {(_Float16)qv.x, (_Float16)qv.y, (_Float16)qv.z, (_Float16)qv.w};
      v4h hk = {(_Float16)kv.x, (_Float16)kv.y, (_Float16)kv.z, (_Float16)kv.w};
      *(v4h*)(sQ + row * LDH + c4) = hq;
      *(v4h*)(sK + row * LDH + c4) = hk;
      sVt[(c4 + 0) * LDH + row] = (_Float16)vv.x;
      sVt[(c4 + 1) * LDH + row] = (_Float16)vv.y;
      sVt[(c4 + 2) * LDH + row] = (_Float16)vv.z;
      sVt[(c4 + 3) * LDH + row] = (_Float16)vv.w;
    } else {
      v4h z = {};
      *(v4h*)(sQ + row * LDH + c4) = z;
      *(v4h*)(sK + row * LDH + c4) = z;
      sVt[(c4 + 0) * LDH + row] = (_Float16)0.f;
      sVt[(c4 + 1) * LDH + row] = (_Float16)0.f;
      sVt[(c4 + 2) * LDH + row] = (_Float16)0.f;
      sVt[(c4 + 3) * LDH + row] = (_Float16)0.f;
    }
  }

  // ---- zero-fill non-dilated output rows of this window (disjoint from computed rows) ----
  if (r > 1) {
    const int nf4 = w * 16;               // float4 count in window
    for (int e = tid; e < nf4; e += 128) {
      const int row = e >> 4;
      const int j   = row - off;
      const bool computed = (j >= 0) && ((j & (r - 1)) == 0) && ((j >> rsh) < mv);
      if (!computed) {
        float4 z = {0.f, 0.f, 0.f, 0.f};
        *(float4*)(out + winBase + (e << 2)) = z;
      }
    }
  }

  __syncthreads();

  // ---- S = (Q K^T) * scale : 16x64 row-block per wave, 4 tiles x 2 k-chunks ----
  const float scale = 0.125f;             // 1/sqrt(64)
  const int row0 = wv * 16;

  v16h aq0 = frag_ld(sQ, row0, 0, lane);
  v16h aq1 = frag_ld(sQ, row0, 32, lane);

  v8f acc[4];
  #pragma unroll
  for (int t = 0; t < 4; ++t) {
    v8f c = {};
    v16h b0 = frag_ld(sK, t * 16, 0, lane);
    v16h b1 = frag_ld(sK, t * 16, 32, lane);
    c = __builtin_amdgcn_wmma_f32_16x16x32_f16(false, aq0, false, b0, (short)0, c, false, false);
    c = __builtin_amdgcn_wmma_f32_16x16x32_f16(false, aq1, false, b1, (short)0, c, false, false);
    acc[t] = c;
  }

  // ---- scale + key-padding mask ----
  #pragma unroll
  for (int t = 0; t < 4; ++t) {
    const bool valid = (t * 16 + (lane & 15)) < mv;
    #pragma unroll
    for (int i = 0; i < 8; ++i)
      acc[t][i] = valid ? acc[t][i] * scale : -1.0e30f;
  }

  // ---- softmax per row: reduce 4 tiles + 16 lanes (xor 1,2,4,8 stays in-half) ----
  float rmax[8], rinv[8];
  #pragma unroll
  for (int i = 0; i < 8; ++i) {
    float m = fmaxf(fmaxf(acc[0][i], acc[1][i]), fmaxf(acc[2][i], acc[3][i]));
    m = fmaxf(m, __shfl_xor(m, 1));
    m = fmaxf(m, __shfl_xor(m, 2));
    m = fmaxf(m, __shfl_xor(m, 4));
    m = fmaxf(m, __shfl_xor(m, 8));
    rmax[i] = m;
  }
  #pragma unroll
  for (int t = 0; t < 4; ++t)
    #pragma unroll
    for (int i = 0; i < 8; ++i)
      acc[t][i] = __expf(acc[t][i] - rmax[i]);
  #pragma unroll
  for (int i = 0; i < 8; ++i) {
    float s = acc[0][i] + acc[1][i] + acc[2][i] + acc[3][i];
    s += __shfl_xor(s, 1);
    s += __shfl_xor(s, 2);
    s += __shfl_xor(s, 4);
    s += __shfl_xor(s, 8);
    rinv[i] = 1.0f / s;
  }

  // ---- P -> LDS (f16), re-fragment for PV ----
  #pragma unroll
  for (int t = 0; t < 4; ++t) {
    const int cn = t * 16 + (lane & 15);
    #pragma unroll
    for (int i = 0; i < 8; ++i) {
      const int rm = row0 + i + ((lane >> 4) << 3);
      sP[rm * LDH + cn] = (_Float16)(acc[t][i] * rinv[i]);
    }
  }
  __syncthreads();

  // ---- O = P V : A from sP, B from transposed V (row = d index) ----
  v16h ap0 = frag_ld(sP, row0, 0, lane);
  v16h ap1 = frag_ld(sP, row0, 32, lane);

  v8f oacc[4];
  #pragma unroll
  for (int t = 0; t < 4; ++t) {
    v8f c = {};
    v16h b0 = frag_ld(sVt, t * 16, 0, lane);
    v16h b1 = frag_ld(sVt, t * 16, 32, lane);
    c = __builtin_amdgcn_wmma_f32_16x16x32_f16(false, ap0, false, b0, (short)0, c, false, false);
    c = __builtin_amdgcn_wmma_f32_16x16x32_f16(false, ap1, false, b1, (short)0, c, false, false);
    oacc[t] = c;
  }

  // ---- scatter O to the dilated rows ----
  #pragma unroll
  for (int i = 0; i < 8; ++i) {
    const int rm = row0 + i + ((lane >> 4) << 3);
    if (rm < mv) {
      const int g = base + rm * (r * 64);
      #pragma unroll
      for (int t = 0; t < 4; ++t)
        out[g + t * 16 + (lane & 15)] = oacc[t][i];
    }
  }
}

extern "C" void kernel_launch(void* const* d_in, const int* in_sizes, int n_in,
                              void* d_out, int out_size, void* d_ws, size_t ws_size,
                              hipStream_t stream) {
  const float* q = (const float*)d_in[0];
  const float* k = (const float*)d_in[1];
  const float* v = (const float*)d_in[2];
  float* out = (float*)d_out;
  (void)in_sizes; (void)n_in; (void)out_size; (void)d_ws; (void)ws_size;

  // 2048 (w=64) + 1024 (w=128) + 512 (w=256) segment-workgroups.
  dilated_attn_f16wmma<<<3584, 128, 0, stream>>>(q, k, v, out);
}
